// SocialAgg_21354577396100
// MI455X (gfx1250) — matrix operand order repaired and checked by reading, not compiled, not run on hardware
//
#include <hip/hip_runtime.h>
#include <hip/hip_bf16.h>

typedef __bf16 bf16_t;
typedef __attribute__((ext_vector_type(16))) __bf16 v16bf;
typedef __attribute__((ext_vector_type(8)))  float  v8f;
typedef __attribute__((ext_vector_type(8)))  unsigned v8u;

#define WAVES 4

// ---------- helpers ----------

__device__ __forceinline__ unsigned pk_bf16(float lo, float hi) {
    unsigned short l = __builtin_bit_cast(unsigned short, (bf16_t)lo);
    unsigned short h = __builtin_bit_cast(unsigned short, (bf16_t)hi);
    return (unsigned)l | ((unsigned)h << 16);
}

// monotone float <-> uint encoding so float max == uint max
__device__ __forceinline__ unsigned enc_f(float f) {
    unsigned b = __float_as_uint(f);
    return (b & 0x80000000u) ? ~b : (b | 0x80000000u);
}
__device__ __forceinline__ float dec_f(unsigned e) {
    unsigned b = (e & 0x80000000u) ? (e & 0x7fffffffu) : ~e;
    return __uint_as_float(b);
}

__device__ __forceinline__ void load8(const float* p, float* r) {
    const float4 x0 = *(const float4*)p;
    const float4 x1 = *(const float4*)(p + 4);
    r[0]=x0.x; r[1]=x0.y; r[2]=x0.z; r[3]=x0.w;
    r[4]=x1.x; r[5]=x1.y; r[6]=x1.z; r[7]=x1.w;
}

// x[k] = (k<64) ? src[k] : dst[k-64]; k is a multiple of 8 so an 8-run never crosses
__device__ __forceinline__ void gather_x(const float* srow, const float* drow, int k, float* r) {
    const float* p = (k < 64) ? (srow + k) : (drow + (k - 64));
    load8(p, r);
}

// A fragment: elements 0..7 <- run0 (K = base..base+7), 8..15 <- run1 (K = base+8..)
__device__ __forceinline__ v16bf pack_a(const float* r0, const float* r1) {
    v16bf a;
#pragma unroll
    for (int j = 0; j < 8; ++j) {
        a[j]     = (bf16_t)r0[j];
        a[8 + j] = (bf16_t)r1[j];
    }
    return a;
}

// B fragment: 8 consecutive packed-pair dwords from an LDS row (32B aligned)
__device__ __forceinline__ v16bf load_b(const unsigned* row, int pairBase) {
    const uint4* p = (const uint4*)(row + pairBase);
    uint4 q0 = p[0], q1 = p[1];
    v8u u;
    u[0]=q0.x; u[1]=q0.y; u[2]=q0.z; u[3]=q0.w;
    u[4]=q1.x; u[5]=q1.y; u[6]=q1.z; u[7]=q1.w;
    return __builtin_bit_cast(v16bf, u);
}

// ---------- K0: init ----------
__global__ void init_kernel(unsigned* __restrict__ menc, float* __restrict__ denom,
                            float* __restrict__ agg, int N) {
    long i = (long)blockIdx.x * blockDim.x + threadIdx.x;
    if (i < N) { menc[i] = 0u; denom[i] = 0.0f; }
    if (i < (long)N * 64) agg[i] = 0.0f;
}

// ---------- K1: per-edge attention MLP (WMMA bf16) ----------
__global__ __launch_bounds__(128) void edge_mlp_kernel(
    const float* __restrict__ user_feat, const float* __restrict__ hi,
    const int* __restrict__ trust, const int* __restrict__ trustee,
    const float* __restrict__ W1, const float* __restrict__ b1,
    const float* __restrict__ W2, const float* __restrict__ b2,
    const float* __restrict__ W3, const float* __restrict__ b3,
    float* __restrict__ scores, int E)
{
    // W packed as bf16 pairs in WMMA-B order: [n][pair], pair p = (W[2p][n], W[2p+1][n])
    __shared__ unsigned w1p[64][64];   // 128x64 -> 16 KB
    __shared__ unsigned w2p[64][32];   // 64x64  ->  8 KB
    __shared__ float w3s[64], b1s[64], b2s[64];
    __shared__ float hbuf[WAVES][16][68];   // padded stride 68 -> conflict-free

    const int tid = threadIdx.x;
    for (int idx = tid; idx < 64 * 64; idx += 128) {
        int n = idx >> 6, p = idx & 63, k = p * 2;
        w1p[n][p] = pk_bf16(W1[k * 64 + n], W1[(k + 1) * 64 + n]);
    }
    for (int idx = tid; idx < 64 * 32; idx += 128) {
        int n = idx >> 5, p = idx & 31, k = p * 2;
        w2p[n][p] = pk_bf16(W2[k * 64 + n], W2[(k + 1) * 64 + n]);
    }
    if (tid < 64) { w3s[tid] = W3[tid]; b1s[tid] = b1[tid]; b2s[tid] = b2[tid]; }
    __syncthreads();

    const int wave = tid >> 5, lane = tid & 31;
    const int half = lane >> 4, mrow = lane & 15;
    const long e0 = ((long)blockIdx.x * WAVES + wave) * 16;
    long em = e0 + mrow; if (em > (long)E - 1) em = (long)E - 1;
    const float* srow = user_feat + (size_t)trust[em] * 64;
    const float* drow = hi + (size_t)trustee[em] * 64;

    // layer-1 A fragments: x = [src || dst], 16 edges x 128
    v16bf a1[4];
#pragma unroll
    for (int c = 0; c < 4; ++c) {
        int k0 = 32 * c + 16 * half;
        float r0[8], r1[8];
        gather_x(srow, drow, k0, r0);
        gather_x(srow, drow, k0 + 8, r1);
        a1[c] = pack_a(r0, r1);
    }

    v8f acc1[4] = {};
#pragma unroll
    for (int nt = 0; nt < 4; ++nt) {
        const unsigned* brow = w1p[nt * 16 + mrow];
#pragma unroll
        for (int c = 0; c < 4; ++c) {
            v16bf b = load_b(brow, 16 * c + 8 * half);
            acc1[nt] = __builtin_amdgcn_wmma_f32_16x16x32_bf16(
                false, a1[c], false, b, (short)0, acc1[nt], false, false);
        }
    }

    // bias + relu, stage h1 (C layout: lane = N, VGPR r = M row (+8 for upper half))
#pragma unroll
    for (int nt = 0; nt < 4; ++nt) {
        int n = nt * 16 + mrow;
        float bias = b1s[n];
#pragma unroll
        for (int r = 0; r < 8; ++r)
            hbuf[wave][8 * half + r][n] = fmaxf(acc1[nt][r] + bias, 0.0f);
    }
    __syncthreads();

    // layer-2 A fragments from staged h1
    v16bf a2[2];
#pragma unroll
    for (int c = 0; c < 2; ++c) {
        int k0 = 32 * c + 16 * half;
        const float* hrow = hbuf[wave][mrow];
        float r0[8], r1[8];
        load8(hrow + k0, r0);
        load8(hrow + k0 + 8, r1);
        a2[c] = pack_a(r0, r1);
    }

    v8f acc2[4] = {};
#pragma unroll
    for (int nt = 0; nt < 4; ++nt) {
        const unsigned* brow = w2p[nt * 16 + mrow];
#pragma unroll
        for (int c = 0; c < 2; ++c) {
            v16bf b = load_b(brow, 16 * c + 8 * half);
            acc2[nt] = __builtin_amdgcn_wmma_f32_16x16x32_bf16(
                false, a2[c], false, b, (short)0, acc2[nt], false, false);
        }
    }
#pragma unroll
    for (int nt = 0; nt < 4; ++nt) {
        int n = nt * 16 + mrow;
        float bias = b2s[n];
#pragma unroll
        for (int r = 0; r < 8; ++r)
            hbuf[wave][8 * half + r][n] = fmaxf(acc2[nt][r] + bias, 0.0f);
    }
    __syncthreads();

    // scores = h2 . w3 + b3 (one edge per lane 0..15)
    if (lane < 16) {
        long e = e0 + lane;
        if (e < E) {
            float s = b3[0];
            const float* hrow = hbuf[wave][lane];
#pragma unroll
            for (int k = 0; k < 64; ++k) s = fmaf(hrow[k], w3s[k], s);
            scores[e] = s;
        }
    }
}

// ---------- K2: segment max over trustee ----------
__global__ void segmax_kernel(const float* __restrict__ scores, const int* __restrict__ trustee,
                              unsigned* __restrict__ menc, int E) {
    long i = (long)blockIdx.x * blockDim.x + threadIdx.x;
    if (i >= E) return;
    atomicMax(&menc[trustee[i]], enc_f(scores[i]));
}

// ---------- K3: exp + segment sum ----------
__global__ void expsum_kernel(const float* __restrict__ scores, const int* __restrict__ trustee,
                              const unsigned* __restrict__ menc, float* __restrict__ eexp,
                              float* __restrict__ denom, int E) {
    long i = (long)blockIdx.x * blockDim.x + threadIdx.x;
    if (i >= E) return;
    int u = trustee[i];
    float ev = __expf(scores[i] - dec_f(menc[u]));
    eexp[i] = ev;
    atomicAdd(&denom[u], ev);
}

// ---------- K4: agg[trustee] += value * hi[trust]  (16 threads/edge x float4) ----------
__global__ void scatter_kernel(const float* __restrict__ hi, const int* __restrict__ trust,
                               const int* __restrict__ trustee, const float* __restrict__ eexp,
                               const float* __restrict__ denom, float* __restrict__ agg, int E) {
    long gid = (long)blockIdx.x * blockDim.x + threadIdx.x;
    if (gid >= (long)E * 16) return;
    int e = (int)(gid >> 4);
    int q = (int)(gid & 15);
    int t = trust[e], u = trustee[e];
    float v = eexp[e] / denom[u];
    const float4 hv = *(const float4*)(hi + (size_t)t * 64 + q * 4);
    float* dst = agg + (size_t)u * 64 + q * 4;
    atomicAdd(dst + 0, v * hv.x);
    atomicAdd(dst + 1, v * hv.y);
    atomicAdd(dst + 2, v * hv.z);
    atomicAdd(dst + 3, v * hv.w);
}

// ---------- K5: hs = relu(agg @ Ww + bw), in-place on agg (WMMA bf16) ----------
__global__ __launch_bounds__(128) void out_gemm_kernel(
    const float* __restrict__ Ww, const float* __restrict__ bw,
    float* __restrict__ agg_hs, int N)
{
    __shared__ unsigned wwp[64][32];
    __shared__ float bws[64];
    const int tid = threadIdx.x;
    for (int idx = tid; idx < 64 * 32; idx += 128) {
        int n = idx >> 5, p = idx & 31, k = p * 2;
        wwp[n][p] = pk_bf16(Ww[k * 64 + n], Ww[(k + 1) * 64 + n]);
    }
    if (tid < 64) bws[tid] = bw[tid];
    __syncthreads();

    const int wave = tid >> 5, lane = tid & 31;
    const int half = lane >> 4, mrow = lane & 15;
    const long row0 = ((long)blockIdx.x * WAVES + wave) * 16;
    long mg = row0 + mrow; if (mg > (long)N - 1) mg = (long)N - 1;
    const float* arow = agg_hs + (size_t)mg * 64;

    v16bf a[2];
#pragma unroll
    for (int c = 0; c < 2; ++c) {
        int k0 = 32 * c + 16 * half;
        float r0[8], r1[8];
        load8(arow + k0, r0);
        load8(arow + k0 + 8, r1);
        a[c] = pack_a(r0, r1);
    }

    v8f acc[4] = {};
#pragma unroll
    for (int nt = 0; nt < 4; ++nt) {
        const unsigned* brow = wwp[nt * 16 + mrow];
#pragma unroll
        for (int c = 0; c < 2; ++c) {
            v16bf b = load_b(brow, 16 * c + 8 * half);
            acc[nt] = __builtin_amdgcn_wmma_f32_16x16x32_bf16(
                false, a[c], false, b, (short)0, acc[nt], false, false);
        }
    }

#pragma unroll
    for (int nt = 0; nt < 4; ++nt) {
        int n = nt * 16 + mrow;
        float bias = bws[n];
#pragma unroll
        for (int r = 0; r < 8; ++r) {
            long node = row0 + 8 * half + r;
            if (node < N) agg_hs[(size_t)node * 64 + n] = fmaxf(acc[nt][r] + bias, 0.0f);
        }
    }
}

// ---------- host launcher ----------
extern "C" void kernel_launch(void* const* d_in, const int* in_sizes, int n_in,
                              void* d_out, int out_size, void* d_ws, size_t ws_size,
                              hipStream_t stream)
{
    const float* user_feat = (const float*)d_in[0];
    const float* hi        = (const float*)d_in[1];
    const int*   trust     = (const int*)d_in[2];
    const int*   trustee   = (const int*)d_in[3];
    const float* W1 = (const float*)d_in[4];
    const float* b1 = (const float*)d_in[5];
    const float* W2 = (const float*)d_in[6];
    const float* b2 = (const float*)d_in[7];
    const float* W3 = (const float*)d_in[8];
    const float* b3 = (const float*)d_in[9];
    const float* Ww = (const float*)d_in[10];
    const float* bw = (const float*)d_in[11];

    const int E = in_sizes[2];
    const int N = in_sizes[0] / 64;

    char* ws = (char*)d_ws;
    size_t off = 0;
    auto take = [&](size_t bytes) {
        size_t o = off;
        off = (off + bytes + 255) & ~(size_t)255;
        return o;
    };
    float*    scores = (float*)(ws + take((size_t)E * 4));
    float*    eexp   = (float*)(ws + take((size_t)E * 4));
    unsigned* menc   = (unsigned*)(ws + take((size_t)N * 4));
    float*    denom  = (float*)(ws + take((size_t)N * 4));
    float*    agg    = (float*)d_out;   // N*64 accumulator, also final output (in-place K5)

    {   // K0
        long total = (long)N * 64;
        init_kernel<<<(int)((total + 255) / 256), 256, 0, stream>>>(menc, denom, agg, N);
    }
    {   // K1
        int tiles = (E + 15) / 16;
        int blocks = (tiles + WAVES - 1) / WAVES;
        edge_mlp_kernel<<<blocks, 128, 0, stream>>>(user_feat, hi, trust, trustee,
                                                    W1, b1, W2, b2, W3, b3, scores, E);
    }
    segmax_kernel<<<(E + 255) / 256, 256, 0, stream>>>(scores, trustee, menc, E);
    expsum_kernel<<<(E + 255) / 256, 256, 0, stream>>>(scores, trustee, menc, eexp, denom, E);
    {   // K4
        long total = (long)E * 16;
        scatter_kernel<<<(int)((total + 255) / 256), 256, 0, stream>>>(hi, trust, trustee,
                                                                       eexp, denom, agg, E);
    }
    {   // K5
        int tiles = (N + 15) / 16;
        out_gemm_kernel<<<(tiles + WAVES - 1) / WAVES, 128, 0, stream>>>(Ww, bw, agg, N);
    }
}
